// BiLSTM_CRF_86157043958403
// MI455X (gfx1250) — compile-verified, matrix operand order checked by reading, not run on hardware
//
#include <hip/hip_runtime.h>
#include <hip/hip_bf16.h>

typedef __attribute__((ext_vector_type(16))) _Float16 v16h;
typedef __attribute__((ext_vector_type(8)))  _Float16 v8h;
typedef __attribute__((ext_vector_type(8)))  float    v8f;

#define B_SZ 128
#define T_SZ 1024
#define E_SZ 256
#define H_SZ 128
#define G_SZ 512   // 4*H
#define K_TAGS 4

// Load a 16-bit WMMA fragment (A or B) from row-major storage, row pointer
// already applied. Lane holds two 8-element K chunks at +0 and +16 halves.
__device__ __forceinline__ v16h load_frag(const _Float16* p) {
  v8h a = *(const v8h*)(p);
  v8h b = *(const v8h*)(p + 16);
  v16h r;
#pragma unroll
  for (int i = 0; i < 8; ++i) { r[i] = a[i]; r[i + 8] = b[i]; }
  return r;
}

__device__ __forceinline__ float sigm(float x) {
  return 1.0f / (1.0f + __expf(-x));
}

// CDNA5 async global->LDS DMA (ASYNCcnt tracked, no VGPR staging).
__device__ __forceinline__ void async_load_b128(unsigned lds_off, const void* gaddr) {
  asm volatile("global_load_async_to_lds_b128 %0, %1, off"
               :: "v"(lds_off), "v"(gaddr) : "memory");
}
__device__ __forceinline__ void wait_async0() {
  asm volatile("s_wait_asynccnt 0" ::: "memory");
}

// ---------------- Kernel 1: embedding gather + f32->f16 ----------------
__global__ void k_embed_gather(const int* __restrict__ sent,
                               const float* __restrict__ emb,
                               _Float16* __restrict__ X) {
  size_t idx = (size_t)blockIdx.x * 256 + threadIdx.x;   // over T*B*E
  int e = (int)(idx & (E_SZ - 1));
  size_t tb = idx >> 8;            // t*B + b
  int b = (int)(tb & (B_SZ - 1));
  int t = (int)(tb >> 7);
  int tok = sent[(size_t)b * T_SZ + t];                  // sentences is [B,T]
  X[idx] = (_Float16)emb[(size_t)tok * E_SZ + e];
}

// ---------------- Kernel 1b: generic f32 -> f16 convert ----------------
__global__ void k_f32_to_f16(const float* __restrict__ src,
                             _Float16* __restrict__ dst, int n) {
  int i = blockIdx.x * 256 + threadIdx.x;
  if (i < n) dst[i] = (_Float16)src[i];
}

// ---------------- Kernel 2: input projection GEMM (WMMA) ----------------
// XP[m, n] = sum_k X[m,k] * Wih[n,k] + bias[n];  M=131072, N=512, K=256
__global__ void k_proj_gemm(const _Float16* __restrict__ X,
                            const _Float16* __restrict__ Wih,
                            const float* __restrict__ bias,
                            _Float16* __restrict__ XP) {
  int lane = threadIdx.x & 31;
  int wv   = threadIdx.x >> 5;     // 0..7 -> N block of 64
  int hi   = lane >> 4;
  int lm   = lane & 15;
  size_t m0 = (size_t)blockIdx.x * 32;

  v8f acc[2][4];
#pragma unroll
  for (int mt = 0; mt < 2; ++mt)
#pragma unroll
    for (int nt = 0; nt < 4; ++nt) acc[mt][nt] = (v8f)(0.0f);

#pragma unroll 2
  for (int kc = 0; kc < E_SZ / 32; ++kc) {
    int kb = kc * 32 + hi * 8;
    v16h a0 = load_frag(X + (m0 + lm) * E_SZ + kb);
    v16h a1 = load_frag(X + (m0 + 16 + lm) * E_SZ + kb);
#pragma unroll
    for (int nt = 0; nt < 4; ++nt) {
      int n = wv * 64 + nt * 16 + lm;
      v16h bf = load_frag(Wih + (size_t)n * E_SZ + kb);
      acc[0][nt] = __builtin_amdgcn_wmma_f32_16x16x32_f16(
          false, a0, false, bf, (short)0, acc[0][nt], false, false);
      acc[1][nt] = __builtin_amdgcn_wmma_f32_16x16x32_f16(
          false, a1, false, bf, (short)0, acc[1][nt], false, false);
    }
  }

#pragma unroll
  for (int mt = 0; mt < 2; ++mt)
#pragma unroll
    for (int nt = 0; nt < 4; ++nt) {
      int n = wv * 64 + nt * 16 + lm;
      float bv = bias[n];
#pragma unroll
      for (int r = 0; r < 8; ++r) {
        size_t m = m0 + mt * 16 + r + hi * 8;
        XP[m * G_SZ + n] = (_Float16)(acc[mt][nt][r] + bv);
      }
    }
}

// ---------------- Kernel 3: recurrent LSTM scan (both directions) ----------------
// grid = (8, 2): x = batch tile of 16 rows, y = direction. block = 256 (8 waves).
// Wave w owns 64 gate columns; its w_hh B-fragments live in REGISTERS for all
// 1024 steps (loop-invariant, 128 VGPRs). h-state A-fragments come from LDS.
// Next step's xp tile (16 KB, contiguous) is prefetched by async LDS DMA and
// double-buffered, overlapping HBM latency with WMMA + activation work.
__global__ void k_lstm_scan(const _Float16* __restrict__ XPf,
                            const _Float16* __restrict__ XPb,
                            const _Float16* __restrict__ WhhF,
                            const _Float16* __restrict__ WhhB,
                            _Float16* __restrict__ Hf,
                            _Float16* __restrict__ Hb) {
  __shared__ _Float16 sH[16 * H_SZ];        // 4 KB  : h state [m][k]
  __shared__ _Float16 sG[16 * G_SZ];        // 16 KB : activated gates [m][n]
  __shared__ _Float16 sXP[2][16 * G_SZ];    // 2x16 KB: double-buffered xp tile

  const int dir = blockIdx.y;
  const _Float16* __restrict__ XP  = dir ? XPb : XPf;
  const _Float16* __restrict__ Whh = dir ? WhhB : WhhF;
  _Float16* __restrict__ Hout      = dir ? Hb : Hf;
  const int b0 = blockIdx.x * 16;

  const int lane = threadIdx.x & 31;
  const int wv   = threadIdx.x >> 5;   // 8 waves; gate = wv>>1 (0=i 1=f 2=g 3=o)
  const int hi   = lane >> 4;
  const int lm   = lane & 15;
  const int gate = wv >> 1;

  // --- loop-invariant w_hh B fragments -> registers (16 x v16h = 128 VGPRs)
  v16h breg[4][4];
#pragma unroll
  for (int nt = 0; nt < 4; ++nt) {
    int n = wv * 64 + nt * 16 + lm;
#pragma unroll
    for (int kc = 0; kc < 4; ++kc)
      breg[nt][kc] = load_frag(Whh + (size_t)n * H_SZ + kc * 32 + hi * 8);
  }

  // zero h state
  for (int i = threadIdx.x; i < 16 * H_SZ; i += 256) sH[i] = (_Float16)0.0f;

  // cell state: thread owns column j for 8 batch rows
  const int j  = threadIdx.x & 127;
  const int mb = (threadIdx.x >> 7) * 8;
  float c[8];
#pragma unroll
  for (int mm = 0; mm < 8; ++mm) c[mm] = 0.0f;

  const unsigned lds_xp0 = (unsigned)(uintptr_t)(&sXP[0][0]);
  const unsigned lds_xp1 = (unsigned)(uintptr_t)(&sXP[1][0]);

  // prefill xp tile for the first timestep
  {
    int t0 = dir ? (T_SZ - 1) : 0;
    const char* base = (const char*)(XP + ((size_t)t0 * B_SZ + b0) * G_SZ);
#pragma unroll
    for (int kchunk = 0; kchunk < 4; ++kchunk) {
      int chunk = threadIdx.x + kchunk * 256;   // 1024 x 16B = 16 KB
      async_load_b128(lds_xp0 + chunk * 16, base + chunk * 16);
    }
    wait_async0();
  }
  __syncthreads();

  int cb = 0;
  for (int step = 0; step < T_SZ; ++step) {
    int t = dir ? (T_SZ - 1 - step) : step;

    // prefetch next step's xp tile into the other buffer (async DMA)
    if (step + 1 < T_SZ) {
      int tn = dir ? (t - 1) : (t + 1);
      const char* base = (const char*)(XP + ((size_t)tn * B_SZ + b0) * G_SZ);
      unsigned ldsn = cb ? lds_xp0 : lds_xp1;
#pragma unroll
      for (int kchunk = 0; kchunk < 4; ++kchunk) {
        int chunk = threadIdx.x + kchunk * 256;
        async_load_b128(ldsn + chunk * 16, base + chunk * 16);
      }
    }

    const _Float16* xp_c = &sXP[cb][0];

    // gates[16, 512] = h @ w_hh^T + xp ; A from LDS, B from registers
    v16h afr[4];
#pragma unroll
    for (int kc = 0; kc < 4; ++kc)
      afr[kc] = load_frag(sH + lm * H_SZ + kc * 32 + hi * 8);

#pragma unroll
    for (int nt = 0; nt < 4; ++nt) {
      int n = wv * 64 + nt * 16 + lm;
      v8f acc = (v8f)(0.0f);
#pragma unroll
      for (int kc = 0; kc < 4; ++kc)
        acc = __builtin_amdgcn_wmma_f32_16x16x32_f16(
            false, afr[kc], false, breg[nt][kc], (short)0, acc, false, false);
#pragma unroll
      for (int r = 0; r < 8; ++r) {
        int m = r + hi * 8;
        float pre = acc[r] + (float)xp_c[m * G_SZ + n];
        float act = (gate == 2) ? tanhf(pre) : sigm(pre);
        sG[m * G_SZ + n] = (_Float16)act;
      }
    }
    __syncthreads();

    // combine: c' = sig(f)*c + sig(i)*tanh(g); h = sig(o)*tanh(c')
#pragma unroll
    for (int mm = 0; mm < 8; ++mm) {
      int m = mb + mm;
      float iv = (float)sG[m * G_SZ + j];
      float fv = (float)sG[m * G_SZ + 128 + j];
      float gv = (float)sG[m * G_SZ + 256 + j];
      float ov = (float)sG[m * G_SZ + 384 + j];
      float cm = fv * c[mm] + iv * gv;
      c[mm] = cm;
      float hv = ov * tanhf(cm);
      sH[m * H_SZ + j] = (_Float16)hv;
      Hout[((size_t)t * B_SZ + b0 + m) * H_SZ + j] = (_Float16)hv;
    }

    wait_async0();       // drain this wave's prefetch before buffer swap
    __syncthreads();
    cb ^= 1;
  }
}

// ---------------- Kernel 4: output projection feats[T,B,4] ----------------
__global__ void k_out_proj(const _Float16* __restrict__ Hf,
                           const _Float16* __restrict__ Hb,
                           const float* __restrict__ Wout,
                           const float* __restrict__ bout,
                           float* __restrict__ feats) {
  __shared__ float sW[K_TAGS * 2 * H_SZ];
  __shared__ float sB[K_TAGS];
  for (int i = threadIdx.x; i < K_TAGS * 2 * H_SZ; i += 256) sW[i] = Wout[i];
  if (threadIdx.x < K_TAGS) sB[threadIdx.x] = bout[threadIdx.x];
  __syncthreads();

  size_t idx = (size_t)blockIdx.x * 256 + threadIdx.x;   // t*B + b
  const _Float16* hf = Hf + idx * H_SZ;
  const _Float16* hb = Hb + idx * H_SZ;
  float a0 = sB[0], a1 = sB[1], a2 = sB[2], a3 = sB[3];
#pragma unroll 4
  for (int e = 0; e < H_SZ; ++e) {
    float x = (float)hf[e];
    a0 += x * sW[0 * 256 + e]; a1 += x * sW[1 * 256 + e];
    a2 += x * sW[2 * 256 + e]; a3 += x * sW[3 * 256 + e];
  }
#pragma unroll 4
  for (int e = 0; e < H_SZ; ++e) {
    float x = (float)hb[e];
    a0 += x * sW[0 * 256 + 128 + e]; a1 += x * sW[1 * 256 + 128 + e];
    a2 += x * sW[2 * 256 + 128 + e]; a3 += x * sW[3 * 256 + 128 + e];
  }
  float4 o; o.x = a0; o.y = a1; o.z = a2; o.w = a3;
  ((float4*)feats)[idx] = o;
}

// ---------------- Kernel 5: Viterbi decode (K=4) ----------------
__global__ void k_viterbi(const float* __restrict__ feats,
                          const float* __restrict__ startv,
                          const float* __restrict__ endv,
                          const float* __restrict__ trans,
                          unsigned char* __restrict__ hist,
                          float* __restrict__ tags) {
  int b = threadIdx.x;   // 128 threads, one per batch row
  float tr[16];
#pragma unroll
  for (int i = 0; i < 16; ++i) tr[i] = trans[i];

  float4 e0 = ((const float4*)feats)[b];
  float s[4] = { startv[0] + e0.x, startv[1] + e0.y,
                 startv[2] + e0.z, startv[3] + e0.w };

  for (int t = 1; t < T_SZ; ++t) {
    float4 em = ((const float4*)feats)[(size_t)t * B_SZ + b];
    float ns[4]; int bp[4];
#pragma unroll
    for (int k = 0; k < 4; ++k) {
      float best = s[0] + tr[0 * 4 + k]; int bj = 0;
#pragma unroll
      for (int jj = 1; jj < 4; ++jj) {
        float v = s[jj] + tr[jj * 4 + k];
        if (v > best) { best = v; bj = jj; }   // strict > keeps first argmax
      }
      ns[k] = best; bp[k] = bj;
    }
    s[0] = ns[0] + em.x; s[1] = ns[1] + em.y;
    s[2] = ns[2] + em.z; s[3] = ns[3] + em.w;
    size_t hidx = ((size_t)(t - 1) * B_SZ + b) * 4;
    hist[hidx + 0] = (unsigned char)bp[0];
    hist[hidx + 1] = (unsigned char)bp[1];
    hist[hidx + 2] = (unsigned char)bp[2];
    hist[hidx + 3] = (unsigned char)bp[3];
  }

  int last = 0; float best = s[0] + endv[0];
#pragma unroll
  for (int k = 1; k < 4; ++k) {
    float v = s[k] + endv[k];
    if (v > best) { best = v; last = k; }
  }
  tags[(size_t)b * T_SZ + (T_SZ - 1)] = (float)last;
  for (int t = T_SZ - 2; t >= 0; --t) {
    last = hist[((size_t)t * B_SZ + b) * 4 + last];
    tags[(size_t)b * T_SZ + t] = (float)last;
  }
}

extern "C" void kernel_launch(void* const* d_in, const int* in_sizes, int n_in,
                              void* d_out, int out_size, void* d_ws, size_t ws_size,
                              hipStream_t stream) {
  (void)in_sizes; (void)n_in; (void)out_size; (void)ws_size;
  const int*   sent   = (const int*)d_in[0];
  const float* emb    = (const float*)d_in[1];
  const float* w_ih_f = (const float*)d_in[2];
  const float* w_hh_f = (const float*)d_in[3];
  const float* b_f    = (const float*)d_in[4];
  const float* w_ih_b = (const float*)d_in[5];
  const float* w_hh_b = (const float*)d_in[6];
  const float* b_b    = (const float*)d_in[7];
  const float* w_out  = (const float*)d_in[8];
  const float* b_out  = (const float*)d_in[9];
  const float* startv = (const float*)d_in[10];
  const float* endv   = (const float*)d_in[11];
  const float* trans  = (const float*)d_in[12];
  float* tags = (float*)d_out;

  char* w = (char*)d_ws;
  auto carve = [&](size_t bytes) -> char* {
    char* p = w; w += (bytes + 255) & ~(size_t)255; return p;
  };
  _Float16* X    = (_Float16*)carve((size_t)T_SZ * B_SZ * E_SZ * 2);
  _Float16* WihF = (_Float16*)carve((size_t)G_SZ * E_SZ * 2);
  _Float16* WihB = (_Float16*)carve((size_t)G_SZ * E_SZ * 2);
  _Float16* WhhF = (_Float16*)carve((size_t)G_SZ * H_SZ * 2);
  _Float16* WhhB = (_Float16*)carve((size_t)G_SZ * H_SZ * 2);
  _Float16* XPf  = (_Float16*)carve((size_t)T_SZ * B_SZ * G_SZ * 2);
  _Float16* XPb  = (_Float16*)carve((size_t)T_SZ * B_SZ * G_SZ * 2);
  _Float16* Hfp  = (_Float16*)carve((size_t)T_SZ * B_SZ * H_SZ * 2);
  _Float16* Hbp  = (_Float16*)carve((size_t)T_SZ * B_SZ * H_SZ * 2);
  float*    feats = (float*)carve((size_t)T_SZ * B_SZ * K_TAGS * 4);
  unsigned char* hist = (unsigned char*)carve((size_t)(T_SZ - 1) * B_SZ * K_TAGS);

  k_embed_gather<<<(T_SZ * B_SZ * E_SZ) / 256, 256, 0, stream>>>(sent, emb, X);
  k_f32_to_f16<<<(G_SZ * E_SZ) / 256, 256, 0, stream>>>(w_ih_f, WihF, G_SZ * E_SZ);
  k_f32_to_f16<<<(G_SZ * E_SZ) / 256, 256, 0, stream>>>(w_ih_b, WihB, G_SZ * E_SZ);
  k_f32_to_f16<<<(G_SZ * H_SZ) / 256, 256, 0, stream>>>(w_hh_f, WhhF, G_SZ * H_SZ);
  k_f32_to_f16<<<(G_SZ * H_SZ) / 256, 256, 0, stream>>>(w_hh_b, WhhB, G_SZ * H_SZ);

  k_proj_gemm<<<(T_SZ * B_SZ) / 32, 256, 0, stream>>>(X, WihF, b_f, XPf);
  k_proj_gemm<<<(T_SZ * B_SZ) / 32, 256, 0, stream>>>(X, WihB, b_b, XPb);

  k_lstm_scan<<<dim3(8, 2), 256, 0, stream>>>(XPf, XPb, WhhF, WhhB, Hfp, Hbp);

  k_out_proj<<<(T_SZ * B_SZ) / 256, 256, 0, stream>>>(Hfp, Hbp, w_out, b_out, feats);

  k_viterbi<<<1, 128, 0, stream>>>(feats, startv, endv, trans, hist, tags);
}
